// MHSAttention_3917010174579
// MI455X (gfx1250) — matrix operand order repaired and checked by reading, not compiled, not run on hardware
//
#include <hip/hip_runtime.h>
#include <hip/hip_bf16.h>
#include <math.h>

// ---------------------------------------------------------------------------
// MHA + RoPE for MI455X (gfx1250), wave32 + v_wmma_f32_16x16x32_bf16.
// Stage 1: f32 -> bf16 conversion of x, qkv_w, o_w (vectorized)
// Stage 2: QKV GEMM (LDS-staged double-buffered B, reg double-buffered A,
//          dscnt-only barrier, branch-free steady loop) + fused RoPE
// Stage 3: flash attention, 1 wave == 16 q rows, KV tiles of 32
// Stage 4: output projection GEMM (same staged loop), f32 out
// ---------------------------------------------------------------------------

#define DEV __device__ __forceinline__

typedef __bf16 bf16;
typedef __attribute__((ext_vector_type(16))) __bf16 v16bf;
typedef __attribute__((ext_vector_type(8)))  __bf16 v8bf;
typedef __attribute__((ext_vector_type(4)))  __bf16 v4bf;
typedef __attribute__((ext_vector_type(8)))  float  v8f;

static constexpr int BATCH = 4;
static constexpr int SEQ   = 2048;
static constexpr int DM    = 1024;
static constexpr int NH    = 16;
static constexpr int DK    = 64;
static constexpr int E3    = 3 * DM;          // 3072
static constexpr int MROWS = BATCH * SEQ;     // 8192

DEV v16bf make16(v8bf lo, v8bf hi) {
    union { v16bf v; v8bf h[2]; } u;
    u.h[0] = lo; u.h[1] = hi;
    return u.v;
}

// A fragment, 16x32 bf16 (ISA 7.12.2): lane m = lane&15 is the row;
// halves e=0..7 -> K = hi*8 + e ; e=8..15 -> K = 16 + hi*8 + (e-8).
DEV v16bf load_a_frag(const bf16* rowp, int lane) {
    const int hi8 = (lane >> 4) << 3;
    v8bf lo = *(const v8bf*)(rowp + hi8);
    v8bf hi = *(const v8bf*)(rowp + 16 + hi8);
    return make16(lo, hi);
}

// B fragment, 32x16 bf16: lane n = lane&15 is the column; same K order as A:
// e=0..7 -> K = hi*8 + e ; e=8..15 -> K = 16 + hi*8 + (e-8). colp -> &Bt[n][k0].
DEV v16bf load_b_frag(const bf16* colp, int lane) {
    const int hi8 = (lane >> 4) << 3;
    v8bf lo = *(const v8bf*)(colp + hi8);
    v8bf hi = *(const v8bf*)(colp + 16 + hi8);
    return make16(lo, hi);
}

DEV v8f wmma_bf16(v16bf a, v16bf b, v8f c) {
    return __builtin_amdgcn_wmma_f32_16x16x32_bf16(
        false, a, false, b, (short)0, c, false, false);
}

DEV v8f vzero8() {
    v8f z;
#pragma unroll
    for (int i = 0; i < 8; ++i) z[i] = 0.0f;
    return z;
}

// Workgroup barrier that only drains DScnt (LDS producer/consumer safety)
// and leaves global loads in flight — unlike __syncthreads(), which emits
// s_wait_loadcnt_dscnt 0x0 and kills the cross-iteration global prefetch.
DEV void lds_barrier() {
    asm volatile("s_wait_dscnt 0x0\n\t"
                 "s_barrier_signal -1\n\t"
                 "s_barrier_wait -1" ::: "memory");
}

// ---------------------------------------------------------------------------
__global__ void __launch_bounds__(256)
cvt_f32_bf16(const float* __restrict__ in, bf16* __restrict__ out, int n) {
    const int n4 = n >> 2;
    const float4* in4 = (const float4*)in;
    v4bf* out4 = (v4bf*)out;
    int i = blockIdx.x * blockDim.x + threadIdx.x;
    const int stride = gridDim.x * blockDim.x;
    for (; i < n4; i += stride) {
        float4 f = in4[i];
        v4bf o;
        o[0] = (bf16)f.x; o[1] = (bf16)f.y; o[2] = (bf16)f.z; o[3] = (bf16)f.w;
        out4[i] = o;
    }
}

// ---------------------------------------------------------------------------
// Staged GEMM inner loop shared by stages 2 and 4.
// Block = 256 thr (8 waves): tile M=128 (16/wave), N=64, K-step 32.
// B strip [64 x 32] staged in LDS (double buffered, one dscnt-barrier/iter);
// A fragment double-buffered in registers. Steady loop is branch-free (last
// iteration peeled); sched_barrier keeps all 8 ds_loads ahead of the WMMAs
// so waits are graduated instead of a full dscnt drain per WMMA.
struct GemmAcc { v8f a[4]; };

DEV GemmAcc gemm_64n(const bf16* __restrict__ arow,   // &A[lane's m-row][0]
                     const bf16* __restrict__ bsrcg,  // &Bt[n0 + cn][ck]
                     bf16* __restrict__ bsh,          // LDS, 2 * 64*32 bf16
                     int tid, int lane, int ln) {
    const int cn = tid >> 2;          // 0..63 : B row this thread stages
    const int ck = (tid & 3) * 8;     // 0,8,16,24
    GemmAcc g;
#pragma unroll
    for (int j = 0; j < 4; ++j) g.a[j] = vzero8();

    // prologue: stage k0=0, preload A frag
    *(v8bf*)(bsh + cn * 32 + ck) = *(const v8bf*)(bsrcg);
    v16bf a_cur = load_a_frag(arow, lane);
    lds_barrier();

    // steady state: 31 branch-free iterations, unconditional prefetch
#pragma unroll 2
    for (int k0 = 0; k0 < DM - 32; k0 += 32) {
        const int buf = (k0 >> 5) & 1;
        v8bf  st_n  = *(const v8bf*)(bsrcg + k0 + 32);
        v16bf a_nxt = load_a_frag(arow + k0 + 32, lane);
        const bf16* bbase = bsh + buf * (64 * 32);
        v16bf b0 = load_b_frag(bbase + (0 * 16 + ln) * 32, lane);
        v16bf b1 = load_b_frag(bbase + (1 * 16 + ln) * 32, lane);
        v16bf b2 = load_b_frag(bbase + (2 * 16 + ln) * 32, lane);
        v16bf b3 = load_b_frag(bbase + (3 * 16 + ln) * 32, lane);
        __builtin_amdgcn_sched_barrier(0);   // loads above, WMMAs below
        g.a[0] = wmma_bf16(a_cur, b0, g.a[0]);
        g.a[1] = wmma_bf16(a_cur, b1, g.a[1]);
        g.a[2] = wmma_bf16(a_cur, b2, g.a[2]);
        g.a[3] = wmma_bf16(a_cur, b3, g.a[3]);
        // st_n's s_wait_loadcnt lands here, after the WMMAs
        *(v8bf*)(bsh + (buf ^ 1) * (64 * 32) + cn * 32 + ck) = st_n;
        lds_barrier();
        a_cur = a_nxt;
    }

    // peeled final iteration (k0 = DM-32, buf = 1): no prefetch, no barrier
    {
        const bf16* bbase = bsh + (((DM - 32) >> 5) & 1) * (64 * 32);
        v16bf b0 = load_b_frag(bbase + (0 * 16 + ln) * 32, lane);
        v16bf b1 = load_b_frag(bbase + (1 * 16 + ln) * 32, lane);
        v16bf b2 = load_b_frag(bbase + (2 * 16 + ln) * 32, lane);
        v16bf b3 = load_b_frag(bbase + (3 * 16 + ln) * 32, lane);
        __builtin_amdgcn_sched_barrier(0);
        g.a[0] = wmma_bf16(a_cur, b0, g.a[0]);
        g.a[1] = wmma_bf16(a_cur, b1, g.a[1]);
        g.a[2] = wmma_bf16(a_cur, b2, g.a[2]);
        g.a[3] = wmma_bf16(a_cur, b3, g.a[3]);
    }
    return g;
}

// ---------------------------------------------------------------------------
// QKV GEMM + RoPE. grid = (E3/64, MROWS/128), block = 256.
__global__ void __launch_bounds__(256)
qkv_rope_kernel(const bf16* __restrict__ xb, const bf16* __restrict__ w,
                const int* __restrict__ pos, bf16* __restrict__ q,
                bf16* __restrict__ k, bf16* __restrict__ vT) {
    __shared__ bf16 bsh[2 * 64 * 32];
    const int tid = threadIdx.x;
    const int wave = tid >> 5, lane = tid & 31;
    const int ln = lane & 15, hi = lane >> 4;
    const int m0 = (blockIdx.y * 8 + wave) * 16;
    const int n0 = blockIdx.x * 64;

    GemmAcc g = gemm_64n(xb + (size_t)(m0 + ln) * DM,
                         w + (size_t)(n0 + (tid >> 2)) * DM + (tid & 3) * 8,
                         bsh, tid, lane, ln);

#pragma unroll
    for (int j = 0; j < 4; ++j) {
        const int ncol  = n0 + j * 16 + ln;   // tpart,h uniform over the wave
        const int tpart = ncol >> 10;         // 0=q 1=k 2=v
        const int rem   = ncol & (DM - 1);
        const int h     = rem >> 6;
        const int d     = rem & (DK - 1);
#pragma unroll
        for (int r = 0; r < 8; ++r) {
            const int row = m0 + r + 8 * hi;
            const int b   = row >> 11;
            const int s   = row & (SEQ - 1);
            float val = g.a[j][r];
            if (tpart == 2) {
                vT[((size_t)(b * NH + h) * DK + d) * SEQ + s] = (bf16)val;
            } else {
                // RoPE pair (2j,2j+1) sits in adjacent lanes
                float partner = __shfl_xor(val, 1);
                const int jp = d >> 1;
                const float freq = __expf(-(float)jp * (9.2103403720f / 32.0f));
                const float ang  = (float)pos[s] * freq;
                float sn, cs;
                __sincosf(ang, &sn, &cs);
                float outv = (d & 1) ? (partner * sn + val * cs)
                                     : (val * cs - partner * sn);
                bf16* dst = (tpart == 0) ? q : k;
                dst[((size_t)(b * NH + h) * SEQ + s) * DK + d] = (bf16)outv;
            }
        }
    }
}

// ---------------------------------------------------------------------------
// Flash attention. grid = 1024 x 256; 1 wave == one 16-row Q tile of one (b,h).
__global__ void __launch_bounds__(256)
flash_attn_kernel(const bf16* __restrict__ q, const bf16* __restrict__ k,
                  const bf16* __restrict__ vT, bf16* __restrict__ ctx) {
    __shared__ bf16 pbuf[8][16 * 32];
    const int wave = threadIdx.x >> 5, lane = threadIdx.x & 31;
    const int ln = lane & 15, hi = lane >> 4;
    const int g  = blockIdx.x * 8 + wave;
    const int bh = g >> 7;
    const int qt = g & 127;
    const int b = bh >> 4, h = bh & 15;
    const int q0 = qt * 16;

    const bf16* qp = q  + (size_t)(b * NH + h) * SEQ * DK;
    const bf16* kp = k  + (size_t)(b * NH + h) * SEQ * DK;
    const bf16* vp = vT + (size_t)(b * NH + h) * DK * SEQ;

    const v16bf qa0 = load_a_frag(qp + (size_t)(q0 + ln) * DK + 0,  lane);
    const v16bf qa1 = load_a_frag(qp + (size_t)(q0 + ln) * DK + 32, lane);

    v8f cacc[4];
#pragma unroll
    for (int c = 0; c < 4; ++c) cacc[c] = vzero8();
    float mx[8], sm[8];
#pragma unroll
    for (int r = 0; r < 8; ++r) { mx[r] = -3.0e38f; sm[r] = 0.0f; }

    bf16* pb = pbuf[wave];

    for (int kv0 = 0; kv0 <= q0 + 15; kv0 += 32) {
        // ---- issue ALL tile loads first (LOADcnt retires in order: score
        //      WMMAs only wait for the K frags; V latency is hidden by the
        //      softmax VALU section) ----
        const bf16* kc0 = kp + (size_t)(kv0 + ln) * DK;
        const bf16* kc1 = kp + (size_t)(kv0 + 16 + ln) * DK;
        v16bf kb00 = load_b_frag(kc0 + 0,  lane);
        v16bf kb01 = load_b_frag(kc0 + 32, lane);
        v16bf kb10 = load_b_frag(kc1 + 0,  lane);
        v16bf kb11 = load_b_frag(kc1 + 32, lane);
        v16bf vb[4];
#pragma unroll
        for (int c = 0; c < 4; ++c)
            vb[c] = load_b_frag(vp + (size_t)(c * 16 + ln) * SEQ + kv0, lane);
        __builtin_amdgcn_sched_barrier(0);   // keep all loads ahead of WMMAs

        // ---- scores: two 16x16 tiles, QK^T over dk=64 ----
        v8f s0 = vzero8(), s1 = vzero8();
        s0 = wmma_bf16(qa0, kb00, s0);
        s0 = wmma_bf16(qa1, kb01, s0);
        s1 = wmma_bf16(qa0, kb10, s1);
        s1 = wmma_bf16(qa1, kb11, s1);

        // ---- online softmax (row stats via wave32 half-shuffles) ----
        float p0[8], p1[8];
#pragma unroll
        for (int r = 0; r < 8; ++r) {
            const int qi  = q0 + r + 8 * hi;
            const int kj0 = kv0 + ln, kj1 = kv0 + 16 + ln;
            float v0 = s0[r] * 0.125f;    // 1/sqrt(64)
            float v1 = s1[r] * 0.125f;
            if (kj0 > qi) v0 = -3.0e38f;
            if (kj1 > qi) v1 = -3.0e38f;
            float t = fmaxf(v0, v1);
            t = fmaxf(t, __shfl_xor(t, 1));
            t = fmaxf(t, __shfl_xor(t, 2));
            t = fmaxf(t, __shfl_xor(t, 4));
            t = fmaxf(t, __shfl_xor(t, 8));
            const float mnew = fmaxf(mx[r], t);
            const float f  = __expf(mx[r] - mnew);
            const float e0 = __expf(v0 - mnew);
            const float e1 = __expf(v1 - mnew);
            float rs = e0 + e1;
            rs += __shfl_xor(rs, 1);
            rs += __shfl_xor(rs, 2);
            rs += __shfl_xor(rs, 4);
            rs += __shfl_xor(rs, 8);
            sm[r] = sm[r] * f + rs;
            mx[r] = mnew;
            p0[r] = e0; p1[r] = e1;
#pragma unroll
            for (int c = 0; c < 4; ++c) cacc[c][r] *= f;
        }

        // ---- relayout P (C-layout f32 -> A-layout bf16) through LDS ----
#pragma unroll
        for (int r = 0; r < 8; ++r) {
            const int m = r + 8 * hi;
            pb[m * 32 + ln]      = (bf16)p0[r];
            pb[m * 32 + 16 + ln] = (bf16)p1[r];
        }
        __builtin_amdgcn_wave_barrier();   // LDS in-order within wave; block
                                           // only compiler reordering
        v16bf pa = load_a_frag(pb + ln * 32, lane);

        // ---- ctx += P @ V (V frags loaded long ago) ----
#pragma unroll
        for (int c = 0; c < 4; ++c) cacc[c] = wmma_bf16(pa, vb[c], cacc[c]);
    }

#pragma unroll
    for (int c = 0; c < 4; ++c) {
#pragma unroll
        for (int r = 0; r < 8; ++r) {
            const int m = r + 8 * hi;
            const float val = cacc[c][r] / sm[r];
            ctx[((size_t)b * SEQ + q0 + m) * DM + h * DK + c * 16 + ln] = (bf16)val;
        }
    }
}

// ---------------------------------------------------------------------------
// Output projection: out[8192,1024] f32 = ctx_bf16 @ o_w^T.
__global__ void __launch_bounds__(256)
out_proj_kernel(const bf16* __restrict__ ctx, const bf16* __restrict__ wo,
                float* __restrict__ out) {
    __shared__ bf16 bsh[2 * 64 * 32];
    const int tid = threadIdx.x;
    const int wave = tid >> 5, lane = tid & 31;
    const int ln = lane & 15, hi = lane >> 4;
    const int m0 = (blockIdx.y * 8 + wave) * 16;
    const int n0 = blockIdx.x * 64;

    GemmAcc g = gemm_64n(ctx + (size_t)(m0 + ln) * DM,
                         wo + (size_t)(n0 + (tid >> 2)) * DM + (tid & 3) * 8,
                         bsh, tid, lane, ln);

#pragma unroll
    for (int j = 0; j < 4; ++j) {
#pragma unroll
        for (int r = 0; r < 8; ++r) {
            out[(size_t)(m0 + r + 8 * hi) * DM + n0 + j * 16 + ln] = g.a[j][r];
        }
    }
}

// ---------------------------------------------------------------------------
extern "C" void kernel_launch(void* const* d_in, const int* in_sizes, int n_in,
                              void* d_out, int out_size, void* d_ws, size_t ws_size,
                              hipStream_t stream) {
    const float* x     = (const float*)d_in[0];
    const int*   tpos  = (const int*)d_in[1];
    const float* qkv_w = (const float*)d_in[2];
    const float* o_w   = (const float*)d_in[3];
    float* out = (float*)d_out;

    bf16* wsb  = (bf16*)d_ws;
    bf16* xb   = wsb;                               // 8192*1024
    bf16* wqkv = xb   + (size_t)MROWS * DM;         // 3072*1024
    bf16* wo   = wqkv + (size_t)E3 * DM;            // 1024*1024
    bf16* qb   = wo   + (size_t)DM * DM;
    bf16* kb   = qb   + (size_t)MROWS * DM;
    bf16* vTb  = kb   + (size_t)MROWS * DM;
    bf16* ctxb = vTb  + (size_t)MROWS * DM;

    cvt_f32_bf16<<<1024, 256, 0, stream>>>(x,     xb,   MROWS * DM);
    cvt_f32_bf16<<<512,  256, 0, stream>>>(qkv_w, wqkv, E3 * DM);
    cvt_f32_bf16<<<256,  256, 0, stream>>>(o_w,   wo,   DM * DM);

    qkv_rope_kernel<<<dim3(E3 / 64, MROWS / 128), 256, 0, stream>>>(
        xb, wqkv, tpos, qb, kb, vTb);

    flash_attn_kernel<<<1024, 256, 0, stream>>>(qb, kb, vTb, ctxb);

    out_proj_kernel<<<dim3(DM / 64, MROWS / 128), 256, 0, stream>>>(
        ctxb, wo, out);
}
